// GatedShortConvBlock_3753801417578
// MI455X (gfx1250) — matrix operand order repaired and checked by reading, not compiled, not run on hardware
//
#include <hip/hip_runtime.h>
#include <hip/hip_bf16.h>

// ---------------------------------------------------------------------------
// GatedShortConvBlock fused pipeline for MI455X (gfx1250, wave32, WMMA+TDM).
//   x:[8,4096,1024] f32 -> LN -> GEMM1(3H) -> silu-gate -> causal dwconv(K=4)
//   -> gate -> GEMM2 -> +bias +residual -> out f32
// GEMMs: v_wmma_f32_16x16x32_f16, A staged in LDS by tensor_load_to_lds
// (double-buffered, TENSORcnt-synchronized), B direct from L2-resident f16
// weights.
// ---------------------------------------------------------------------------

typedef __attribute__((ext_vector_type(16))) _Float16 v16h;
typedef __attribute__((ext_vector_type(8)))  float    v8f;
typedef __attribute__((ext_vector_type(4)))  unsigned int u32x4;
typedef __attribute__((ext_vector_type(8)))  int      i32x8;
typedef __attribute__((ext_vector_type(4)))  int      i32x4;

#define DIMD   1024
#define HID    2048
#define SEQ    4096
#define BATCH  8
#define MROWS  (BATCH * SEQ)   // 32768 tokens
#define KCONV  4
#define KCHUNK 256             // K-chunk staged per TDM transfer (32 x 256 f16)

// The TDM builtin writes LDS through an address carried inside the
// descriptor, not through a pointer the compiler can see. Without this,
// clang proves the shared tile is never stored and folds its loads to
// poison (observed: ds_loads deleted, WMMA A-operands collapsed). This
// escape + memory clobber makes the tile externally-modifiable.
__device__ __forceinline__ void lds_escape(const void* p) {
  asm volatile("" :: "r"(p) : "memory");
}

// --------------------------- TDM 2D tile load ------------------------------
// D# per CDNA5 ISA: group0 = {count, lds_addr,
// global_addr, type=2}; group1 = {data_size=2B, tensor dims, tile dims,
// tensor_dim0_stride}. Loads tile_rows x tile_k f16 elements, rows packed
// contiguously in LDS (row pitch = tile_k elements).
// 6-arg builtin form: (u32x4 g0, i32x8 g1, i32x4 g2, i32x4 g3, i32x8 g4, cpol)
__device__ __forceinline__ void tdm_load_2d(unsigned lds_byte_off,
                                            const _Float16* gptr,
                                            unsigned tile_k,
                                            unsigned tile_rows,
                                            unsigned row_stride_elems) {
  const unsigned long long ga = (unsigned long long)(uintptr_t)gptr;
  const unsigned long long st = (unsigned long long)row_stride_elems;
  u32x4 g0;
  g0[0] = 1u;                                                // count=1
  g0[1] = lds_byte_off;                                      // lds_addr
  g0[2] = (unsigned)(ga & 0xffffffffu);                      // gaddr[31:0]
  g0[3] = (unsigned)((ga >> 32) & 0x01ffffffu) | (2u << 30); // gaddr[56:32]|type=2
  i32x8 g1;
  g1[0] = (int)(1u << 16);                                   // data_size=1 (2B)
  g1[1] = (int)((row_stride_elems & 0xffffu) << 16);         // tensor_dim0[15:0]
  g1[2] = (int)(((row_stride_elems >> 16) & 0xffffu) |
                ((tile_rows & 0xffffu) << 16));              // t_dim0[31:16]|t_dim1[15:0]
  g1[3] = (int)(((tile_rows >> 16) & 0xffffu) |
                ((tile_k & 0xffffu) << 16));                 // t_dim1[31:16]|tile_dim0
  g1[4] = (int)(tile_rows & 0xffffu);                        // tile_dim1 (tile_dim2=0)
  g1[5] = (int)(st & 0xffffffffu);                           // dim0_stride[31:0]
  g1[6] = (int)((st >> 32) & 0xffffu);                       // dim0_stride[47:32]
  g1[7] = 0;
  const i32x4 z4 = {0, 0, 0, 0};
  const i32x8 z8 = {0, 0, 0, 0, 0, 0, 0, 0};
  __builtin_amdgcn_tensor_load_to_lds(g0, g1, z4, z4, z8, 0);
}

// ----------------------------- fragment loads ------------------------------
// A-matrix 16x32 f16 (ISA 7.12.2): lane l (r=l&15, h=l>>4) holds row r,
//   VGPR0-3: K = kk + h*8 + [0..7], VGPR4-7: K = kk + 16 + h*8 + [0..7].
__device__ __forceinline__ v16h load_frag_a_p(const _Float16* T, int pitch,
                                              int kk, int lane) {
  const int r = lane & 15, h = lane >> 4;
  const _Float16* p = T + (size_t)r * pitch + kk + h * 8;
  union { uint4 q[2]; v16h v; } u;
  u.q[0] = *(const uint4*)(p);
  u.q[1] = *(const uint4*)(p + 16);
  return u.v;
}

// B-matrix 32x16 f16: lane l (c=l&15, h=l>>4) holds column n0+c,
// K = k0 + h*8 + [0..7] and k0 + 16 + h*8 + [0..7]. W stored [N][K] row-major.
__device__ __forceinline__ v16h load_frag_b(const _Float16* __restrict__ W,
                                            int ldk, int n0, int k0, int lane) {
  const int c = lane & 15, h = lane >> 4;
  const _Float16* p = W + (size_t)(n0 + c) * ldk + k0 + h * 8;
  union { uint4 q[2]; v16h v; } u;
  u.q[0] = *(const uint4*)(p);
  u.q[1] = *(const uint4*)(p + 16);
  return u.v;
}

__device__ __forceinline__ v8f wmma_f16(v16h a, v16h b, v8f c) {
  return __builtin_amdgcn_wmma_f32_16x16x32_f16(
      false, a, false, b, (short)0, c, false, false);
}

__device__ __forceinline__ float sigmoidf_fast(float x) {
  return 1.0f / (1.0f + __expf(-x));
}

// ----------------------------- cast f32 -> f16 -----------------------------
__global__ __launch_bounds__(256) void cast_f32_to_f16(
    const float* __restrict__ in, _Float16* __restrict__ out, int n4) {
  int i = blockIdx.x * 256 + threadIdx.x;
  if (i >= n4) return;
  float4 v = *(const float4*)(in + (size_t)i * 4);
  union { uint2 q; _Float16 h[4]; } o;
  o.h[0] = (_Float16)v.x; o.h[1] = (_Float16)v.y;
  o.h[2] = (_Float16)v.z; o.h[3] = (_Float16)v.w;
  *(uint2*)(out + (size_t)i * 4) = o.q;
}

// ----------------------------- LayerNorm -> f16 ----------------------------
__global__ __launch_bounds__(256) void ln_to_f16(
    const float* __restrict__ x, const float* __restrict__ gamma,
    const float* __restrict__ beta, _Float16* __restrict__ out) {
  __shared__ float s1[256];
  __shared__ float s2[256];
  const int row = blockIdx.x;
  const int t = threadIdx.x;
  const float* xr = x + (size_t)row * DIMD;
  float4 v = *(const float4*)(xr + t * 4);
  float sum = v.x + v.y + v.z + v.w;
  float sq  = v.x * v.x + v.y * v.y + v.z * v.z + v.w * v.w;
  s1[t] = sum; s2[t] = sq;
  __syncthreads();
  for (int s = 128; s > 0; s >>= 1) {
    if (t < s) { s1[t] += s1[t + s]; s2[t] += s2[t + s]; }
    __syncthreads();
  }
  const float mu  = s1[0] * (1.0f / DIMD);
  const float var = s2[0] * (1.0f / DIMD) - mu * mu;
  const float rs  = rsqrtf(var + 1e-5f);
  union { uint2 q; _Float16 h[4]; } o;
  const int base = t * 4;
  o.h[0] = (_Float16)((v.x - mu) * rs * gamma[base + 0] + beta[base + 0]);
  o.h[1] = (_Float16)((v.y - mu) * rs * gamma[base + 1] + beta[base + 1]);
  o.h[2] = (_Float16)((v.z - mu) * rs * gamma[base + 2] + beta[base + 2]);
  o.h[3] = (_Float16)((v.w - mu) * rs * gamma[base + 3] + beta[base + 3]);
  *(uint2*)(out + (size_t)row * DIMD + base) = o.q;
}

// ------------------------------- GEMM1 -------------------------------------
// proj = h @ w_in^T with fused first gate. Block: 8 waves, 32 rows x 128 cols.
// A strip (32 x K) staged in LDS via double-buffered TDM; 6 WMMAs per k-step.
__global__ __launch_bounds__(256) void gemm1_gate(
    const _Float16* __restrict__ hA, const _Float16* __restrict__ wB,
    const float* __restrict__ b_in,
    _Float16* __restrict__ vOut, _Float16* __restrict__ gOut) {
  __shared__ _Float16 atile[2][32][KCHUNK];   // 2 x 16KB, LDS offset 0
  const int lane = threadIdx.x & 31;
  const int wave = threadIdx.x >> 5;
  const int n0 = (blockIdx.x * 8 + wave) * 16;  // column tile within H
  const int m0 = blockIdx.y * 32;               // row tile (shared by block)
  const int K = DIMD;
  const int nchunk = K / KCHUNK;

  if (wave == 0)
    tdm_load_2d(0, hA + (size_t)m0 * K, KCHUNK, 32, K);

  v8f aB0 = {0,0,0,0,0,0,0,0}, aB1 = {0,0,0,0,0,0,0,0};
  v8f aC0 = {0,0,0,0,0,0,0,0}, aC1 = {0,0,0,0,0,0,0,0};
  v8f aH0 = {0,0,0,0,0,0,0,0}, aH1 = {0,0,0,0,0,0,0,0};

  for (int c = 0; c < nchunk; ++c) {
    if (wave == 0) __builtin_amdgcn_s_wait_tensorcnt(0);
    __syncthreads();                      // chunk c resident for all waves
    lds_escape(&atile[0][0][0]);          // tile was written by the TDM
    if (wave == 0 && c + 1 < nchunk)      // overlap DMA of c+1 with compute
      tdm_load_2d(((c + 1) & 1) * (32 * KCHUNK * 2),
                  hA + (size_t)m0 * K + (c + 1) * KCHUNK, KCHUNK, 32, K);
    const _Float16* abuf = &atile[c & 1][0][0];
#pragma unroll 2
    for (int kk = 0; kk < KCHUNK; kk += 32) {
      const int kg = c * KCHUNK + kk;
      v16h a0 = load_frag_a_p(abuf,               KCHUNK, kk, lane);
      v16h a1 = load_frag_a_p(abuf + 16 * KCHUNK, KCHUNK, kk, lane);
      v16h bB = load_frag_b(wB, K, n0,           kg, lane);
      v16h bC = load_frag_b(wB, K, HID + n0,     kg, lane);
      v16h bH = load_frag_b(wB, K, 2 * HID + n0, kg, lane);
      aB0 = wmma_f16(a0, bB, aB0);  aB1 = wmma_f16(a1, bB, aB1);
      aC0 = wmma_f16(a0, bC, aC0);  aC1 = wmma_f16(a1, bC, aC1);
      aH0 = wmma_f16(a0, bH, aH0);  aH1 = wmma_f16(a1, bH, aH1);
    }
    __syncthreads();                      // done reading before overwrite
  }

  // Epilogue: C layout -> lane (col = lane&15, hs = lane>>4), VGPR r holds
  // row m0 + 16*sub + 8*hs + r.
  const int col = lane & 15;
  const int hs  = lane >> 4;
  const int n   = n0 + col;
  const float bg = b_in[n];
  const float bc = b_in[HID + n];
  const float bh = b_in[2 * HID + n];
#pragma unroll
  for (int sub = 0; sub < 2; ++sub) {
    v8f* pB = sub ? &aB1 : &aB0;
    v8f* pC = sub ? &aC1 : &aC0;
    v8f* pH = sub ? &aH1 : &aH0;
#pragma unroll
    for (int r = 0; r < 8; ++r) {
      const int row = m0 + sub * 16 + hs * 8 + r;
      const float gb = (*pB)[r] + bg;
      const float gc = (*pC)[r] + bc;
      const float hd = (*pH)[r] + bh;
      const size_t o = (size_t)row * HID + n;
      vOut[o] = (_Float16)(hd * sigmoidf_fast(gb));
      gOut[o] = (_Float16)sigmoidf_fast(gc);
    }
  }
}

// --------------------- depthwise causal conv + gate ------------------------
__global__ __launch_bounds__(256) void dwconv_gate(
    const _Float16* __restrict__ v16, const _Float16* __restrict__ g16,
    const float* __restrict__ conv_w, const float* __restrict__ conv_b,
    _Float16* __restrict__ u16) {
  const int row = blockIdx.x;           // b*SEQ + t
  const int t   = row & (SEQ - 1);
  const int c0  = threadIdx.x * 8;      // 256 * 8 = 2048 channels
  union H8 { uint4 q; _Float16 h[8]; };

  H8 taps[KCONV];
#pragma unroll
  for (int j = 0; j < KCONV; ++j) {
    const int dt = t - (KCONV - 1) + j;
    if (dt >= 0) {
      taps[j].q = *(const uint4*)(v16 + (size_t)(row - (KCONV - 1) + j) * HID + c0);
    } else {
      taps[j].q = make_uint4(0u, 0u, 0u, 0u);
    }
  }
  H8 gate; gate.q = *(const uint4*)(g16 + (size_t)row * HID + c0);

  H8 out;
#pragma unroll
  for (int ch = 0; ch < 8; ++ch) {
    const int c = c0 + ch;
    float acc = conv_b[c];
    const float4 w = *(const float4*)(conv_w + (size_t)c * KCONV);
    acc += (float)taps[0].h[ch] * w.x;
    acc += (float)taps[1].h[ch] * w.y;
    acc += (float)taps[2].h[ch] * w.z;
    acc += (float)taps[3].h[ch] * w.w;
    out.h[ch] = (_Float16)(acc * (float)gate.h[ch]);
  }
  *(uint4*)(u16 + (size_t)row * HID + c0) = out.q;
}

// ------------------------------- GEMM2 -------------------------------------
// out = u @ w_out^T + b_out + residual. Same TDM-staged A strip, K = 2048.
__global__ __launch_bounds__(256) void gemm2_residual(
    const _Float16* __restrict__ uA, const _Float16* __restrict__ wB,
    const float* __restrict__ b_out, const float* __restrict__ resid,
    float* __restrict__ out) {
  __shared__ _Float16 atile[2][32][KCHUNK];   // 2 x 16KB, LDS offset 0
  const int lane = threadIdx.x & 31;
  const int wave = threadIdx.x >> 5;
  const int n0 = (blockIdx.x * 8 + wave) * 16;   // column tile within D
  const int m0 = blockIdx.y * 32;
  const int K = HID;
  const int nchunk = K / KCHUNK;

  if (wave == 0)
    tdm_load_2d(0, uA + (size_t)m0 * K, KCHUNK, 32, K);

  v8f a0c = {0,0,0,0,0,0,0,0}, a1c = {0,0,0,0,0,0,0,0};
  for (int c = 0; c < nchunk; ++c) {
    if (wave == 0) __builtin_amdgcn_s_wait_tensorcnt(0);
    __syncthreads();
    lds_escape(&atile[0][0][0]);
    if (wave == 0 && c + 1 < nchunk)
      tdm_load_2d(((c + 1) & 1) * (32 * KCHUNK * 2),
                  uA + (size_t)m0 * K + (c + 1) * KCHUNK, KCHUNK, 32, K);
    const _Float16* abuf = &atile[c & 1][0][0];
#pragma unroll 2
    for (int kk = 0; kk < KCHUNK; kk += 32) {
      const int kg = c * KCHUNK + kk;
      v16h a0 = load_frag_a_p(abuf,               KCHUNK, kk, lane);
      v16h a1 = load_frag_a_p(abuf + 16 * KCHUNK, KCHUNK, kk, lane);
      v16h b  = load_frag_b(wB, K, n0, kg, lane);
      a0c = wmma_f16(a0, b, a0c);
      a1c = wmma_f16(a1, b, a1c);
    }
    __syncthreads();
  }

  const int col = lane & 15;
  const int hs  = lane >> 4;
  const int n   = n0 + col;
  const float bn = b_out[n];
#pragma unroll
  for (int sub = 0; sub < 2; ++sub) {
    v8f* pA = sub ? &a1c : &a0c;
#pragma unroll
    for (int r = 0; r < 8; ++r) {
      const int row = m0 + sub * 16 + hs * 8 + r;
      const size_t o = (size_t)row * DIMD + n;
      out[o] = (*pA)[r] + bn + resid[o];
    }
  }
}

// ------------------------------ launcher -----------------------------------
extern "C" void kernel_launch(void* const* d_in, const int* in_sizes, int n_in,
                              void* d_out, int out_size, void* d_ws, size_t ws_size,
                              hipStream_t stream) {
  const float* x      = (const float*)d_in[0];   // [8,4096,1024]
  const float* ln_g   = (const float*)d_in[1];
  const float* ln_b   = (const float*)d_in[2];
  const float* w_in   = (const float*)d_in[3];   // [3H, D]
  const float* b_in   = (const float*)d_in[4];   // [3H]
  const float* conv_w = (const float*)d_in[5];   // [H,1,4]
  const float* conv_b = (const float*)d_in[6];   // [H]
  const float* w_out  = (const float*)d_in[7];   // [D, H]
  const float* b_out  = (const float*)d_in[8];   // [D]
  float* out = (float*)d_out;

  // Workspace carve (f16 buffers)
  _Float16* h16     = (_Float16*)d_ws;                       // 32768x1024
  _Float16* w_in16  = h16    + (size_t)MROWS * DIMD;         // 6144x1024
  _Float16* w_out16 = w_in16 + (size_t)3 * HID * DIMD;       // 1024x2048
  _Float16* v16     = w_out16 + (size_t)DIMD * HID;          // 32768x2048
  _Float16* g16     = v16    + (size_t)MROWS * HID;          // 32768x2048
  _Float16* u16     = g16    + (size_t)MROWS * HID;          // 32768x2048

  // 1) weight casts
  {
    const int n4_in  = (3 * HID * DIMD) / 4;
    const int n4_out = (DIMD * HID) / 4;
    cast_f32_to_f16<<<(n4_in  + 255) / 256, 256, 0, stream>>>(w_in,  w_in16,  n4_in);
    cast_f32_to_f16<<<(n4_out + 255) / 256, 256, 0, stream>>>(w_out, w_out16, n4_out);
  }

  // 2) LayerNorm -> f16
  ln_to_f16<<<MROWS, 256, 0, stream>>>(x, ln_g, ln_b, h16);

  // 3) GEMM1 + first gate
  gemm1_gate<<<dim3(HID / 16 / 8, MROWS / 32), 256, 0, stream>>>(
      h16, w_in16, b_in, v16, g16);

  // 4) depthwise causal conv + second gate
  dwconv_gate<<<MROWS, 256, 0, stream>>>(v16, g16, conv_w, conv_b, u16);

  // 5) GEMM2 + bias + residual
  gemm2_residual<<<dim3(DIMD / 16 / 8, MROWS / 32), 256, 0, stream>>>(
      u16, w_out16, b_out, x, out);
}
